// SmallMoEModel_76192719831313
// MI455X (gfx1250) — matrix operand order repaired and checked by reading, not compiled, not run on hardware
//
#include <hip/hip_runtime.h>
#include <hip/hip_bf16.h>

// ---------------- model dims ----------------
#define Hh   1024
#define Ss   1024
#define Bb   4
#define Tt   (Bb * Ss)       // 4096 tokens
#define Ll   2
#define Ee   8
#define NHn  4
#define HDn  (Hh / NHn)      // 256
#define Vv   32000

// ---------------- GEMM tiling ----------------
#define MT 128
#define NT 64
#define KT 32
#define LPAD 8               // ushort pad -> row stride 40 ushorts = 80B (16B aligned)

typedef __attribute__((ext_vector_type(16))) __bf16 bf16x16;
typedef __attribute__((ext_vector_type(8)))  __bf16 bf16x8;
typedef __attribute__((ext_vector_type(8)))  float  f32x8;
typedef long long ll;

#if defined(__gfx1250__) && __has_builtin(__builtin_amdgcn_global_load_async_to_lds_b128) && __has_builtin(__builtin_amdgcn_s_wait_asynccnt)
#define USE_ASYNC_LDS 1
typedef int v4i_async __attribute__((__vector_size__(4 * sizeof(int))));
typedef __attribute__((address_space(1))) v4i_async* g_v4i_ptr;
typedef __attribute__((address_space(3))) v4i_async* l_v4i_ptr;
#endif

__device__ __forceinline__ unsigned short f2bf(float f) {
  unsigned int u = __float_as_uint(f);
  return (unsigned short)((u + 0x7fffu + ((u >> 16) & 1u)) >> 16);  // RNE
}

// ============================================================================
// Generic batched GEMM: C[z](M,N) = act(alpha * A[z](M,K) @ B[z] + bias) * rowscale
//  A : fp32 row-major (lda), staged to LDS with fused fp32->bf16 convert
//  Bw: bf16, K-major [N,K] (ldb) -> contiguous 16B/lane chunks; loaded with
//      GLOBAL_LOAD_ASYNC_TO_LDS_B128 when available (ASYNCcnt pipeline)
//  batch z decomposed (z/nb2, z%nb2), two strides per operand
//  Block 256 thr = 8 waves (wave32); wave tile 32x32 -> 4 x v_wmma_f32_16x16x32_bf16
//  Double-buffered LDS (2-deep software pipeline).
// ============================================================================
__global__ __launch_bounds__(256)
void gemm_wmma_bf16(const float* __restrict__ A, ll lda, ll sA1, ll sA2,
                    const unsigned short* __restrict__ Bw, ll ldb, ll sB1, ll sB2,
                    float* __restrict__ C, ll ldc, ll sC1, ll sC2,
                    const float* __restrict__ bias, const float* __restrict__ rowscale,
                    float alpha, int act, int accum, int K, int nb2)
{
  __shared__ unsigned short As[2][MT][KT + LPAD];
  __shared__ unsigned short Bs[2][NT][KT + LPAD];

  const int z  = blockIdx.z;
  const int b1 = z / nb2, b2 = z % nb2;
  A  += (ll)b1 * sA1 + (ll)b2 * sA2;
  Bw += (ll)b1 * sB1 + (ll)b2 * sB2;
  C  += (ll)b1 * sC1 + (ll)b2 * sC2;

  const int m0   = blockIdx.y * MT;
  const int n0   = blockIdx.x * NT;
  const int tid  = threadIdx.x;
  const int lane = tid & 31;
  const int wv   = tid >> 5;
  const int wmB  = (wv >> 1) * 32;   // wave row base: 0,32,64,96
  const int wnB  = (wv & 1) * 32;    // wave col base: 0,32

  // staging assignments
  const int sar = tid >> 1;          // A row 0..127
  const int sac = (tid & 1) * 16;    // A col 0/16
  const int sbn = tid >> 2;          // B row 0..63
  const int sbk = (tid & 3) * 8;     // B col 0/8/16/24  (16B chunk)

  // fragment addressing (ISA 7.12.2): A lanes 0-15 K{0-7,16-23}, 16-31 K{8-15,24-31}
  const int al  = lane & 15;
  const int ah  = lane >> 4;
  const int akb = ah * 8;
  const int bkb = ah * 8;            // B lanes 0-15 K{0-7,16-23}, 16-31 K{8-15,24-31}

  f32x8 acc[2][2];
#pragma unroll
  for (int i = 0; i < 2; ++i)
#pragma unroll
    for (int j = 0; j < 2; ++j)
#pragma unroll
      for (int r = 0; r < 8; ++r) acc[i][j][r] = 0.f;

  union U8 { unsigned short s[8]; uint4 v; };
  union Frag { bf16x16 v; bf16x8 h[2]; };

  auto stageA = [&](int buf, int k0) {
    const float* ap = A + (ll)(m0 + sar) * lda + k0 + sac;
    const float4 v0 = *(const float4*)(ap + 0);
    const float4 v1 = *(const float4*)(ap + 4);
    const float4 v2 = *(const float4*)(ap + 8);
    const float4 v3 = *(const float4*)(ap + 12);
    U8 u0, u1;
    u0.s[0] = f2bf(v0.x); u0.s[1] = f2bf(v0.y); u0.s[2] = f2bf(v0.z); u0.s[3] = f2bf(v0.w);
    u0.s[4] = f2bf(v1.x); u0.s[5] = f2bf(v1.y); u0.s[6] = f2bf(v1.z); u0.s[7] = f2bf(v1.w);
    u1.s[0] = f2bf(v2.x); u1.s[1] = f2bf(v2.y); u1.s[2] = f2bf(v2.z); u1.s[3] = f2bf(v2.w);
    u1.s[4] = f2bf(v3.x); u1.s[5] = f2bf(v3.y); u1.s[6] = f2bf(v3.z); u1.s[7] = f2bf(v3.w);
    *(uint4*)&As[buf][sar][sac]     = u0.v;
    *(uint4*)&As[buf][sar][sac + 8] = u1.v;
    if (k0 + KT < K) __builtin_prefetch(ap + KT, 0, 1);
  };

  auto stageB = [&](int buf, int k0) {
    const unsigned short* bp = Bw + (ll)(n0 + sbn) * ldb + k0 + sbk;
    unsigned short* lp = &Bs[buf][sbn][sbk];
#if defined(USE_ASYNC_LDS)
    __builtin_amdgcn_global_load_async_to_lds_b128((g_v4i_ptr)bp, (l_v4i_ptr)lp, 0, 0);
#else
    *(uint4*)lp = *(const uint4*)bp;
#endif
  };

  stageB(0, 0);
  stageA(0, 0);
#if defined(USE_ASYNC_LDS)
  __builtin_amdgcn_s_wait_asynccnt(0);
#endif
  __syncthreads();

  int buf = 0;
  for (int k0 = 0; k0 < K; k0 += KT) {
    const int nk = k0 + KT;
    if (nk < K) { stageB(buf ^ 1, nk); stageA(buf ^ 1, nk); }

    Frag a0, a1, b0, b1;
    a0.h[0] = *(const bf16x8*)&As[buf][wmB + al][akb];
    a0.h[1] = *(const bf16x8*)&As[buf][wmB + al][akb + 16];
    a1.h[0] = *(const bf16x8*)&As[buf][wmB + 16 + al][akb];
    a1.h[1] = *(const bf16x8*)&As[buf][wmB + 16 + al][akb + 16];
    b0.h[0] = *(const bf16x8*)&Bs[buf][wnB + al][bkb];
    b0.h[1] = *(const bf16x8*)&Bs[buf][wnB + al][bkb + 16];
    b1.h[0] = *(const bf16x8*)&Bs[buf][wnB + 16 + al][bkb];
    b1.h[1] = *(const bf16x8*)&Bs[buf][wnB + 16 + al][bkb + 16];

    acc[0][0] = __builtin_amdgcn_wmma_f32_16x16x32_bf16(false, a0.v, false, b0.v,
                                                        (short)0, acc[0][0], false, false);
    acc[0][1] = __builtin_amdgcn_wmma_f32_16x16x32_bf16(false, a0.v, false, b1.v,
                                                        (short)0, acc[0][1], false, false);
    acc[1][0] = __builtin_amdgcn_wmma_f32_16x16x32_bf16(false, a1.v, false, b0.v,
                                                        (short)0, acc[1][0], false, false);
    acc[1][1] = __builtin_amdgcn_wmma_f32_16x16x32_bf16(false, a1.v, false, b1.v,
                                                        (short)0, acc[1][1], false, false);

#if defined(USE_ASYNC_LDS)
    if (nk < K) __builtin_amdgcn_s_wait_asynccnt(0);
#endif
    __syncthreads();
    buf ^= 1;
  }

  // epilogue: C layout (ISA): VGPR r -> M = r + (lane>=16 ? 8 : 0), N = lane&15
#pragma unroll
  for (int ri = 0; ri < 2; ++ri) {
#pragma unroll
    for (int ci = 0; ci < 2; ++ci) {
      const int col = n0 + wnB + ci * 16 + al;
      const int rb  = m0 + wmB + ri * 16 + ah * 8;
      const float bval = bias ? bias[col] : 0.f;
#pragma unroll
      for (int r = 0; r < 8; ++r) {
        const int row = rb + r;
        float v = alpha * acc[ri][ci][r] + bval;
        if (act == 1) v = 0.5f * v * (1.f + erff(v * 0.70710678118f));   // exact GELU
        else if (act == 2) v = fmaxf(v, 0.f);                            // ReLU
        if (rowscale) v *= rowscale[row];
        float* cp = &C[(ll)row * ldc + col];
        if (accum) *cp += v; else *cp = v;
      }
    }
  }
}

// ============================================================================
// conversion kernels (run once per launch; pure-bandwidth)
// ============================================================================
// elementwise fp32 -> bf16 (n multiple of 2048)
__global__ void cvt_copy_k(const float* __restrict__ src, unsigned short* __restrict__ dst, ll n) {
  const ll i = ((ll)blockIdx.x * 256 + threadIdx.x) * 8;
  if (i + 8 > n) return;
  const float4 v0 = *(const float4*)(src + i);
  const float4 v1 = *(const float4*)(src + i + 4);
  union { unsigned short s[8]; uint4 v; } u;
  u.s[0] = f2bf(v0.x); u.s[1] = f2bf(v0.y); u.s[2] = f2bf(v0.z); u.s[3] = f2bf(v0.w);
  u.s[4] = f2bf(v1.x); u.s[5] = f2bf(v1.y); u.s[6] = f2bf(v1.z); u.s[7] = f2bf(v1.w);
  *(uint4*)(dst + i) = u.v;
}

// tiled transpose+convert: dst[N,K] (bf16, lddst) = src[K,N] (fp32, ldsrc)
// batch z decomposed (z/nb2, z%nb2); grid(N/32, K/32, batch), block(32,8)
__global__ void transpose_cvt_k(const float* __restrict__ src, ll ldsrc, ll sS1, ll sS2,
                                unsigned short* __restrict__ dst, ll lddst, ll sD1, ll sD2,
                                int nb2) {
  const int z = blockIdx.z;
  src += (ll)(z / nb2) * sS1 + (ll)(z % nb2) * sS2;
  dst += (ll)(z / nb2) * sD1 + (ll)(z % nb2) * sD2;
  __shared__ float tile[32][33];
  const int nb_ = blockIdx.x * 32, kb = blockIdx.y * 32;
#pragma unroll
  for (int j = threadIdx.y; j < 32; j += 8)
    tile[j][threadIdx.x] = src[(ll)(kb + j) * ldsrc + nb_ + threadIdx.x];
  __syncthreads();
#pragma unroll
  for (int j = threadIdx.y; j < 32; j += 8)
    dst[(ll)(nb_ + j) * lddst + kb + threadIdx.x] = f2bf(tile[threadIdx.x][j]);
}

// ---------------- small kernels ----------------
__global__ void embed_gather_k(const int* __restrict__ ids, const float* __restrict__ emb,
                               float* __restrict__ x) {
  const int t = blockIdx.x;
  const ll src = (ll)ids[t] * Hh;
#pragma unroll
  for (int i = 0; i < Hh / 256; ++i) {
    const int c = threadIdx.x + i * 256;
    x[(ll)t * Hh + c] = emb[src + c];
  }
}

// x = LN(x + r) * g + b   (one block per row, H=1024)
__global__ void ln_res_k(float* __restrict__ x, const float* __restrict__ r,
                         const float* __restrict__ g, const float* __restrict__ b) {
  const int row = blockIdx.x, tid = threadIdx.x;
  __shared__ float red[256];
  float v[4]; float s = 0.f;
#pragma unroll
  for (int i = 0; i < 4; ++i) {
    const int c = tid + i * 256;
    v[i] = x[(ll)row * Hh + c] + r[(ll)row * Hh + c];
    s += v[i];
  }
  red[tid] = s; __syncthreads();
  for (int st = 128; st > 0; st >>= 1) { if (tid < st) red[tid] += red[tid + st]; __syncthreads(); }
  const float mean = red[0] * (1.f / Hh);
  __syncthreads();
  s = 0.f;
#pragma unroll
  for (int i = 0; i < 4; ++i) { const float d = v[i] - mean; s += d * d; }
  red[tid] = s; __syncthreads();
  for (int st = 128; st > 0; st >>= 1) { if (tid < st) red[tid] += red[tid + st]; __syncthreads(); }
  const float rstd = rsqrtf(red[0] * (1.f / Hh) + 1e-5f);
#pragma unroll
  for (int i = 0; i < 4; ++i) {
    const int c = tid + i * 256;
    x[(ll)row * Hh + c] = (v[i] - mean) * rstd * g[c] + b[c];
  }
}

// in-place softmax over rows of width 1024 (one block per row)
__global__ void softmax1024_k(float* __restrict__ p) {
  const ll row = blockIdx.x;
  const int tid = threadIdx.x;
  __shared__ float red[256];
  float v[4]; float m = -3.4e38f;
#pragma unroll
  for (int i = 0; i < 4; ++i) { v[i] = p[row * 1024 + tid + i * 256]; m = fmaxf(m, v[i]); }
  red[tid] = m; __syncthreads();
  for (int st = 128; st > 0; st >>= 1) { if (tid < st) red[tid] = fmaxf(red[tid], red[tid + st]); __syncthreads(); }
  m = red[0]; __syncthreads();
  float s = 0.f;
#pragma unroll
  for (int i = 0; i < 4; ++i) { v[i] = expf(v[i] - m); s += v[i]; }
  red[tid] = s; __syncthreads();
  for (int st = 128; st > 0; st >>= 1) { if (tid < st) red[tid] += red[tid + st]; __syncthreads(); }
  const float inv = 1.f / red[0];
#pragma unroll
  for (int i = 0; i < 4; ++i) p[row * 1024 + tid + i * 256] = v[i] * inv;
}

// top-2 gate -> binary masks[Ee][Tt]
__global__ void gate_top2_k(const float* __restrict__ x, const float* __restrict__ gw,
                            const float* __restrict__ gb, float* __restrict__ masks) {
  const int t = blockIdx.x, tid = threadIdx.x;
  float acc[Ee];
#pragma unroll
  for (int e = 0; e < Ee; ++e) acc[e] = 0.f;
  for (int i = tid; i < Hh; i += 256) {
    const float xv = x[(ll)t * Hh + i];
#pragma unroll
    for (int e = 0; e < Ee; ++e) acc[e] += xv * gw[i * Ee + e];
  }
  __shared__ float red[Ee][256];
#pragma unroll
  for (int e = 0; e < Ee; ++e) red[e][tid] = acc[e];
  __syncthreads();
  for (int st = 128; st > 0; st >>= 1) {
    if (tid < st)
#pragma unroll
      for (int e = 0; e < Ee; ++e) red[e][tid] += red[e][tid + st];
    __syncthreads();
  }
  if (tid == 0) {
    float lg[Ee];
#pragma unroll
    for (int e = 0; e < Ee; ++e) lg[e] = red[e][0] + gb[e];
    int i1 = 0;
    for (int e = 1; e < Ee; ++e) if (lg[e] > lg[i1]) i1 = e;
    int i2 = -1;
    for (int e = 0; e < Ee; ++e) if (e != i1 && (i2 < 0 || lg[e] > lg[i2])) i2 = e;
#pragma unroll
    for (int e = 0; e < Ee; ++e) masks[(ll)e * Tt + t] = (e == i1 || e == i2) ? 1.f : 0.f;
  }
}

// ---------------- host side ----------------
static inline void launch_gemm(hipStream_t st,
    const float* A, ll lda, ll sA1, ll sA2,
    const unsigned short* B, ll ldb, ll sB1, ll sB2,
    float* C, ll ldc, ll sC1, ll sC2,
    const float* bias, const float* rowscale, float alpha, int act, int accum,
    int M, int N, int K, int nbz, int nb2) {
  dim3 grid(N / NT, M / MT, nbz);
  gemm_wmma_bf16<<<grid, 256, 0, st>>>(A, lda, sA1, sA2, B, ldb, sB1, sB2,
                                       C, ldc, sC1, sC2, bias, rowscale, alpha, act, accum, K, nb2);
}

extern "C" void kernel_launch(void* const* d_in, const int* in_sizes, int n_in,
                              void* d_out, int out_size, void* d_ws, size_t ws_size,
                              hipStream_t stream) {
  const int*   ids    = (const int*)  d_in[0];
  const float* embed  = (const float*)d_in[1];
  const float* qkvw   = (const float*)d_in[2];
  const float* qkvb   = (const float*)d_in[3];
  const float* outw   = (const float*)d_in[4];
  const float* outb   = (const float*)d_in[5];
  const float* ln1g   = (const float*)d_in[6];
  const float* ln1b   = (const float*)d_in[7];
  const float* ffw1   = (const float*)d_in[8];
  const float* ffb1   = (const float*)d_in[9];
  const float* ffw2   = (const float*)d_in[10];
  const float* ffb2   = (const float*)d_in[11];
  const float* ln2g   = (const float*)d_in[12];
  const float* ln2b   = (const float*)d_in[13];
  const float* gatew  = (const float*)d_in[14];
  const float* gateb  = (const float*)d_in[15];
  const float* ew1    = (const float*)d_in[16];
  const float* eb1    = (const float*)d_in[17];
  const float* ew2    = (const float*)d_in[18];
  const float* eb2    = (const float*)d_in[19];
  const float* lmw    = (const float*)d_in[20];
  const float* lmb    = (const float*)d_in[21];
  float* out = (float*)d_out;

  // ---- fp32 workspace ----
  float* ws   = (float*)d_ws;
  float* X0   = ws;                                 // [4096,1024]
  float* X1   = X0 + (ll)Tt * Hh;                   // [4096,1024]
  float* AO   = X1 + (ll)Tt * Hh;                   // [4096,1024]
  float* TMP  = AO + (ll)Tt * Hh;                   // up to [4096,4096]
  float* SC   = TMP + (ll)Tt * 4 * Hh;              // [B,NH,S,S]
  float* MASK = SC + (ll)Bb * NHn * Ss * Ss;        // [E,T]
  // ---- bf16 (K-major) workspace ----
  unsigned short* WB    = (unsigned short*)(MASK + (ll)Ee * Tt);
  unsigned short* QKVB  = WB;                       // [T,3H]
  unsigned short* VT    = QKVB + (ll)Tt * 3 * Hh;   // [B,NH,HD,S]
  unsigned short* WQKV  = VT + (ll)Bb * NHn * HDn * Ss;   // [L,3H,H]
  unsigned short* WOUT  = WQKV + (ll)Ll * 3 * Hh * Hh;    // [L,H,H]
  unsigned short* WFF1  = WOUT + (ll)Ll * Hh * Hh;        // [L,4H,H]
  unsigned short* WFF2  = WFF1 + (ll)Ll * 4 * Hh * Hh;    // [L,H,4H]
  unsigned short* WE1   = WFF2 + (ll)Ll * Hh * 4 * Hh;    // [L,E,2H,H]
  unsigned short* WE2   = WE1 + (ll)Ll * Ee * 2 * Hh * Hh;// [L,E,H,2H]
  unsigned short* WLM   = WE2 + (ll)Ll * Ee * Hh * 2 * Hh;// [V,H]

  // ---- one-time (per launch) weight conversion to bf16 K-major ----
  cvt_copy_k<<<(int)((ll)Ll * 3 * Hh * Hh / 2048), 256, 0, stream>>>(qkvw, WQKV, (ll)Ll * 3 * Hh * Hh);
  cvt_copy_k<<<(int)((ll)Ll * Hh * Hh / 2048), 256, 0, stream>>>(outw, WOUT, (ll)Ll * Hh * Hh);
  dim3 tb(32, 8);
  transpose_cvt_k<<<dim3(4 * Hh / 32, Hh / 32, Ll), tb, 0, stream>>>(
      ffw1, 4 * Hh, (ll)Hh * 4 * Hh, 0, WFF1, Hh, (ll)4 * Hh * Hh, 0, 1);
  transpose_cvt_k<<<dim3(Hh / 32, 4 * Hh / 32, Ll), tb, 0, stream>>>(
      ffw2, Hh, (ll)4 * Hh * Hh, 0, WFF2, 4 * Hh, (ll)Hh * 4 * Hh, 0, 1);
  transpose_cvt_k<<<dim3(2 * Hh / 32, Hh / 32, Ll * Ee), tb, 0, stream>>>(
      ew1, 2 * Hh, (ll)Hh * 2 * Hh, 0, WE1, Hh, (ll)2 * Hh * Hh, 0, 1);
  transpose_cvt_k<<<dim3(Hh / 32, 2 * Hh / 32, Ll * Ee), tb, 0, stream>>>(
      ew2, Hh, (ll)2 * Hh * Hh, 0, WE2, 2 * Hh, (ll)Hh * 2 * Hh, 0, 1);
  transpose_cvt_k<<<dim3(Vv / 32, Hh / 32, 1), tb, 0, stream>>>(
      lmw, Vv, 0, 0, WLM, Hh, 0, 0, 1);

  embed_gather_k<<<Tt, 256, 0, stream>>>(ids, embed, X0);

  float* xin = X0;
  float* xalt = X1;
  for (int l = 0; l < Ll; ++l) {
    // qkv = x @ Wqkv^T + b : [4096, 3072]  (fp32 result kept for q)
    launch_gemm(stream, xin, Hh, 0, 0, WQKV + (ll)l * 3 * Hh * Hh, Hh, 0, 0,
                TMP, 3 * Hh, 0, 0, qkvb + (ll)l * 3 * Hh, nullptr, 1.f, 0, 0,
                Tt, 3 * Hh, Hh, 1, 1);

    // bf16 mirror of qkv (K operand for scores), and V^T (K-major for attn@v)
    cvt_copy_k<<<(int)((ll)Tt * 3 * Hh / 2048), 256, 0, stream>>>(TMP, QKVB, (ll)Tt * 3 * Hh);
    transpose_cvt_k<<<dim3(HDn / 32, Ss / 32, Bb * NHn), tb, 0, stream>>>(
        TMP + 2 * Hh, 3 * Hh, (ll)Ss * 3 * Hh, HDn,
        VT, Ss, (ll)NHn * HDn * Ss, (ll)HDn * Ss, NHn);

    // scores[b,h] = q k^T / 16 : per (b,h) 1024x1024, K=256
    launch_gemm(stream, TMP, 3 * Hh, (ll)Ss * 3 * Hh, HDn,
                QKVB + Hh, 3 * Hh, (ll)Ss * 3 * Hh, HDn,
                SC, Ss, (ll)NHn * Ss * Ss, (ll)Ss * Ss,
                nullptr, nullptr, 0.0625f, 0, 0,
                Ss, Ss, HDn, Bb * NHn, NHn);

    softmax1024_k<<<Bb * NHn * Ss, 256, 0, stream>>>(SC);

    // ao = attn @ v : per (b,h) M=1024,N=256,K=1024 (B = V^T, K-major)
    launch_gemm(stream, SC, Ss, (ll)NHn * Ss * Ss, (ll)Ss * Ss,
                VT, Ss, (ll)NHn * HDn * Ss, (ll)HDn * Ss,
                AO, Hh, (ll)Ss * Hh, HDn,
                nullptr, nullptr, 1.f, 0, 0,
                Ss, HDn, Ss, Bb * NHn, NHn);

    // out proj
    launch_gemm(stream, AO, Hh, 0, 0, WOUT + (ll)l * Hh * Hh, Hh, 0, 0,
                xalt, Hh, 0, 0, outb + (ll)l * Hh, nullptr, 1.f, 0, 0,
                Tt, Hh, Hh, 1, 1);

    ln_res_k<<<Tt, 256, 0, stream>>>(xin, xalt, ln1g + (ll)l * Hh, ln1b + (ll)l * Hh);

    // FFN
    launch_gemm(stream, xin, Hh, 0, 0, WFF1 + (ll)l * 4 * Hh * Hh, Hh, 0, 0,
                TMP, 4 * Hh, 0, 0, ffb1 + (ll)l * 4 * Hh, nullptr, 1.f, /*gelu*/1, 0,
                Tt, 4 * Hh, Hh, 1, 1);
    launch_gemm(stream, TMP, 4 * Hh, 0, 0, WFF2 + (ll)l * Hh * 4 * Hh, 4 * Hh, 0, 0,
                xalt, Hh, 0, 0, ffb2 + (ll)l * Hh, nullptr, 1.f, 0, 0,
                Tt, Hh, 4 * Hh, 1, 1);

    ln_res_k<<<Tt, 256, 0, stream>>>(xin, xalt, ln2g + (ll)l * Hh, ln2b + (ll)l * Hh);

    gate_top2_k<<<Tt, 256, 0, stream>>>(xin, gatew + (ll)l * Hh * Ee,
                                        gateb + (ll)l * Ee, MASK);

    // dense experts; binary-mask combine fused into w2 epilogue
    for (int e = 0; e < Ee; ++e) {
      const ll le = (ll)l * Ee + e;
      launch_gemm(stream, xin, Hh, 0, 0, WE1 + le * 2 * Hh * Hh, Hh, 0, 0,
                  TMP, 2 * Hh, 0, 0, eb1 + le * 2 * Hh, nullptr, 1.f, /*relu*/2, 0,
                  Tt, 2 * Hh, Hh, 1, 1);
      launch_gemm(stream, TMP, 2 * Hh, 0, 0, WE2 + le * Hh * 2 * Hh, 2 * Hh, 0, 0,
                  xalt, Hh, 0, 0, eb2 + le * Hh, MASK + (ll)e * Tt, 1.f, 0, (e > 0),
                  Tt, Hh, 2 * Hh, 1, 1);
    }

    float* t = xin; xin = xalt; xalt = t;  // x <- MoE output
  }

  // logits = x @ lm_head + b : [4096, 32000]
  launch_gemm(stream, xin, Hh, 0, 0, WLM, Hh, 0, 0,
              out, Vv, 0, 0, lmb, nullptr, 1.f, 0, 0,
              Tt, Vv, Hh, 1, 1);
}
